// GroupCA_43971875176868
// MI455X (gfx1250) — matrix operand order repaired and checked by reading, not compiled, not run on hardware
//
#include <hip/hip_runtime.h>
#include <hip/hip_bf16.h>
#include <math.h>

// ---------------------------------------------------------------------------
// CDNA5 (gfx1250) wave32 WMMA types
// ---------------------------------------------------------------------------
typedef __attribute__((ext_vector_type(16))) __bf16 v16bf;
typedef __attribute__((ext_vector_type(8)))  float  v8f;
typedef int v4i32 __attribute__((vector_size(16)));

union Frag { uint4 u[2]; v16bf v; };

__device__ __forceinline__ v8f wmma_bf16(v16bf a, v16bf b, v8f c) {
  // D(16x16 f32) = A(16x32 bf16) * B(32x16 bf16) + C
  return __builtin_amdgcn_wmma_f32_16x16x32_bf16(false, a, false, b, (short)0, c,
                                                 false, false);
}

__device__ __forceinline__ float gelu_f(float v) {
  return 0.5f * v * (1.0f + erff(v * 0.70710678118654752f));
}

// ---------------------------------------------------------------------------
// Optional CDNA5 async global->LDS staging (ASYNCcnt-tracked DMA path).
// Builtin signature (from hipcc diagnostics): (int4 AS1*, int4 AS3*, Imm, Imm).
// Falls back to synchronous b128 copies if the builtin is unavailable.
// ---------------------------------------------------------------------------
#if defined(__gfx1250__) && __has_builtin(__builtin_amdgcn_global_load_async_to_lds_b128)
#define USE_ASYNC_LDS 1
__device__ __forceinline__ void async_cp16(const __bf16* g, __bf16* l) {
  __builtin_amdgcn_global_load_async_to_lds_b128(
      (__attribute__((address_space(1))) v4i32*)(g),
      (__attribute__((address_space(3))) v4i32*)(l), 0, 0);
}
#endif

__device__ __forceinline__ void wait_async_lds() {
#ifdef USE_ASYNC_LDS
#if __has_builtin(__builtin_amdgcn_s_wait_asynccnt)
  __builtin_amdgcn_s_wait_asynccnt(0);
#else
  asm volatile("s_wait_asynccnt 0x0" ::: "memory");
#endif
#endif
}

// ---------------------------------------------------------------------------
// fp32 -> bf16 weight conversion
// ---------------------------------------------------------------------------
__global__ void k_f2b(const float* __restrict__ s, __bf16* __restrict__ d, int n) {
  int i = blockIdx.x * 256 + threadIdx.x;
  if (i < n) d[i] = (__bf16)s[i];
}

// ---------------------------------------------------------------------------
// LayerNorm over C=512. MODE 0: bf16 row-major out. MODE 1: fp32 (B,C,H*W) out.
// One block (256 thr) per row of (B*N, C).
// ---------------------------------------------------------------------------
template <int MODE>
__global__ __launch_bounds__(256) void k_layernorm(const float* __restrict__ x,
                                                   const float* __restrict__ w,
                                                   const float* __restrict__ b,
                                                   void* __restrict__ out) {
  const int C = 512;
  int row = blockIdx.x, tid = threadIdx.x;
  const float* xr = x + (size_t)row * C;
  float v0 = xr[tid], v1 = xr[tid + 256];
  __shared__ float red[256];
  red[tid] = v0 + v1;
  __syncthreads();
  for (int s = 128; s > 0; s >>= 1) {
    if (tid < s) red[tid] += red[tid + s];
    __syncthreads();
  }
  float mean = red[0] * (1.0f / C);
  __syncthreads();
  float d0 = v0 - mean, d1 = v1 - mean;
  red[tid] = d0 * d0 + d1 * d1;
  __syncthreads();
  for (int s = 128; s > 0; s >>= 1) {
    if (tid < s) red[tid] += red[tid + s];
    __syncthreads();
  }
  float rstd = rsqrtf(red[0] * (1.0f / C) + 1e-5f);
  float y0 = d0 * rstd * w[tid] + b[tid];
  float y1 = d1 * rstd * w[tid + 256] + b[tid + 256];
  if (MODE == 0) {
    __bf16* o = (__bf16*)out + (size_t)row * C;
    o[tid] = (__bf16)y0;
    o[tid + 256] = (__bf16)y1;
  } else {
    int bb = row >> 10, n = row & 1023;  // N = 1024 = H*W
    float* o = (float*)out;
    o[((size_t)(bb * C + tid)) * 1024 + n] = y0;
    o[((size_t)(bb * C + tid + 256)) * 1024 + n] = y1;
  }
}

// ---------------------------------------------------------------------------
// Generic NT bf16 GEMM: C[M,N] = act(A[M,K] * Bw[N,K]^T + bias) + res
// Block tile 128x128x32, 256 threads = 8 waves (2x4), wave tile 64x32.
// Double-buffered LDS staging; next k-tile staged (async DMA when available)
// while the current tile feeds the WMMA pipe.
// ---------------------------------------------------------------------------
template <bool BIAS, bool ACT, bool RES, bool OUTF, bool OUTB>
__global__ __launch_bounds__(256) void k_gemm_nt(
    const __bf16* __restrict__ A, const __bf16* __restrict__ Bw,
    const float* __restrict__ bias, const float* __restrict__ res,
    float* __restrict__ Cf, __bf16* __restrict__ Cb, int M, int N, int K) {
  __shared__ __bf16 As[2][128][40];  // 80B rows -> 16B aligned b128 LDS ops
  __shared__ __bf16 Bs[2][128][40];
  int tid = threadIdx.x, lane = tid & 31, wave = tid >> 5;
  int l15 = lane & 15, hi = lane >> 4;
  int wm = wave >> 2, wn = wave & 3;
  int m0 = blockIdx.y << 7, n0 = blockIdx.x << 7;
  int lr = tid >> 2, lc = (tid & 3) << 3;
  const __bf16* pa0 = A + (size_t)(m0 + lr) * K + lc;
  const __bf16* pa1 = A + (size_t)(m0 + lr + 64) * K + lc;
  const __bf16* pb0 = Bw + (size_t)(n0 + lr) * K + lc;
  const __bf16* pb1 = Bw + (size_t)(n0 + lr + 64) * K + lc;

  auto stage = [&](int buf, int k0) {
#ifdef USE_ASYNC_LDS
    async_cp16(pa0 + k0, &As[buf][lr][lc]);
    async_cp16(pa1 + k0, &As[buf][lr + 64][lc]);
    async_cp16(pb0 + k0, &Bs[buf][lr][lc]);
    async_cp16(pb1 + k0, &Bs[buf][lr + 64][lc]);
#else
    *(uint4*)&As[buf][lr][lc]      = *(const uint4*)(pa0 + k0);
    *(uint4*)&As[buf][lr + 64][lc] = *(const uint4*)(pa1 + k0);
    *(uint4*)&Bs[buf][lr][lc]      = *(const uint4*)(pb0 + k0);
    *(uint4*)&Bs[buf][lr + 64][lc] = *(const uint4*)(pb1 + k0);
#endif
  };

  v8f acc[4][2];
#pragma unroll
  for (int i = 0; i < 4; i++)
#pragma unroll
    for (int j = 0; j < 2; j++)
#pragma unroll
      for (int r = 0; r < 8; r++) acc[i][j][r] = 0.0f;

  stage(0, 0);
  wait_async_lds();
  __syncthreads();

  for (int k0 = 0; k0 < K; k0 += 32) {
    int cur = (k0 >> 5) & 1;
    if (k0 + 32 < K) stage(cur ^ 1, k0 + 32);  // overlap with WMMA below
#ifndef USE_ASYNC_LDS
    if (k0 + 64 < K) {  // emits global_prefetch for the tile after next
      __builtin_prefetch(pa0 + k0 + 64, 0, 3);
      __builtin_prefetch(pb0 + k0 + 64, 0, 3);
    }
#endif
    Frag a[4], bfr[2];
#pragma unroll
    for (int fm = 0; fm < 4; fm++) {
      int m = (wm << 6) + (fm << 4) + l15;
      a[fm].u[0] = *(const uint4*)&As[cur][m][hi << 3];
      a[fm].u[1] = *(const uint4*)&As[cur][m][16 + (hi << 3)];
    }
#pragma unroll
    for (int fn = 0; fn < 2; fn++) {
      int n = (wn << 5) + (fn << 4) + l15;
      bfr[fn].u[0] = *(const uint4*)&Bs[cur][n][hi << 3];
      bfr[fn].u[1] = *(const uint4*)&Bs[cur][n][16 + (hi << 3)];
    }
#pragma unroll
    for (int fm = 0; fm < 4; fm++)
#pragma unroll
      for (int fn = 0; fn < 2; fn++)
        acc[fm][fn] = wmma_bf16(a[fm].v, bfr[fn].v, acc[fm][fn]);
    wait_async_lds();
    __syncthreads();
  }

#pragma unroll
  for (int fm = 0; fm < 4; fm++) {
#pragma unroll
    for (int fn = 0; fn < 2; fn++) {
      int gn = n0 + (wn << 5) + (fn << 4) + l15;
      float bv = BIAS ? bias[gn] : 0.0f;
#pragma unroll
      for (int r = 0; r < 8; r++) {
        int gm = m0 + (wm << 6) + (fm << 4) + r + (hi << 3);
        float v = acc[fm][fn][r] + bv;
        if (ACT) v = gelu_f(v);
        size_t oi = (size_t)gm * N + gn;
        if (RES) v += res[oi];
        if (OUTF) Cf[oi] = v;
        if (OUTB) Cb[oi] = (__bf16)v;
      }
    }
  }
}

// ---------------------------------------------------------------------------
// Split qkv -> l2-normalized q,k in (b*h, d, N) bf16. Block = one (b,h,dd) row.
// ---------------------------------------------------------------------------
__global__ __launch_bounds__(128) void k_qk_split(const float* __restrict__ qkv,
                                                  __bf16* __restrict__ qh,
                                                  __bf16* __restrict__ kh) {
  int idx = blockIdx.x;  // bh*64 + dd
  int dd = idx & 63, bh = idx >> 6;
  int b = bh >> 3, h = bh & 7;
  int tid = threadIdx.x;
  const float* base = qkv + (size_t)b * 1024 * 1536 + h * 64 + dd;
  float qv[8], kv[8];
  float sq = 0.f, sk = 0.f;
#pragma unroll
  for (int i = 0; i < 8; i++) {
    int n = tid + (i << 7);
    float q = base[(size_t)n * 1536];
    float k = base[(size_t)n * 1536 + 512];
    qv[i] = q; kv[i] = k;
    sq += q * q; sk += k * k;
  }
  __shared__ float rq[128], rk[128];
  rq[tid] = sq; rk[tid] = sk;
  __syncthreads();
  for (int s = 64; s > 0; s >>= 1) {
    if (tid < s) { rq[tid] += rq[tid + s]; rk[tid] += rk[tid + s]; }
    __syncthreads();
  }
  float scq = 1.0f / fmaxf(sqrtf(rq[0]), 1e-12f);
  float sck = 1.0f / fmaxf(sqrtf(rk[0]), 1e-12f);
  __bf16* qo = qh + (size_t)idx * 1024;
  __bf16* ko = kh + (size_t)idx * 1024;
#pragma unroll
  for (int i = 0; i < 8; i++) {
    int n = tid + (i << 7);
    qo[n] = (__bf16)(qv[i] * scq);
    ko[n] = (__bf16)(kv[i] * sck);
  }
}

// v slice of qkv -> bf16 (B*N, C) with C = h*64+e (token rows, K-contiguous)
__global__ void k_v_split(const float* __restrict__ qkv, __bf16* __restrict__ vb) {
  int i = blockIdx.x * 256 + threadIdx.x;  // 16384*512
  int col = i & 511, row = i >> 9;
  vb[i] = (__bf16)qkv[(size_t)row * 1536 + 1024 + col];
}

// ---------------------------------------------------------------------------
// Channel cross-covariance attention core. One block per (b,h).
// Phase 1: S = q k^T (64x64, K=1024) via WMMA, softmax(temperature*S).
// Phase 2: O = P V (64x1024, K=64) via WMMA -> bf16 (B,N,C) layout.
// ---------------------------------------------------------------------------
__global__ __launch_bounds__(256) void k_attn(const __bf16* __restrict__ qh,
                                              const __bf16* __restrict__ kh,
                                              const __bf16* __restrict__ vb,
                                              const float* __restrict__ temp,
                                              __bf16* __restrict__ ob) {
  __shared__ float Sf[64][65];
  __shared__ __bf16 Pb[64][72];  // 144B rows -> 16B aligned
  int bh = blockIdx.x, b = bh >> 3, h = bh & 7;
  int tid = threadIdx.x, lane = tid & 31, wave = tid >> 5;
  int l15 = lane & 15, hi = lane >> 4;
  int wm = wave >> 2, wn = wave & 3;  // 2 x 4 -> wave tile 32x16
  const __bf16* qbase = qh + (size_t)bh * 64 * 1024;
  const __bf16* kbase = kh + (size_t)bh * 64 * 1024;

  v8f s[2];
#pragma unroll
  for (int i = 0; i < 2; i++)
#pragma unroll
    for (int r = 0; r < 8; r++) s[i][r] = 0.0f;

  for (int k0 = 0; k0 < 1024; k0 += 32) {
    Frag a0, a1, bf;
    int m_ = (wm << 5) + l15;
    a0.u[0] = *(const uint4*)(qbase + (size_t)m_ * 1024 + k0 + (hi << 3));
    a0.u[1] = *(const uint4*)(qbase + (size_t)m_ * 1024 + k0 + 16 + (hi << 3));
    a1.u[0] = *(const uint4*)(qbase + (size_t)(m_ + 16) * 1024 + k0 + (hi << 3));
    a1.u[1] = *(const uint4*)(qbase + (size_t)(m_ + 16) * 1024 + k0 + 16 + (hi << 3));
    int n_ = (wn << 4) + l15;
    bf.u[0] = *(const uint4*)(kbase + (size_t)n_ * 1024 + k0 + (hi << 3));
    bf.u[1] = *(const uint4*)(kbase + (size_t)n_ * 1024 + k0 + 16 + (hi << 3));
    s[0] = wmma_bf16(a0.v, bf.v, s[0]);
    s[1] = wmma_bf16(a1.v, bf.v, s[1]);
  }
  float tv = temp[h];
#pragma unroll
  for (int fm = 0; fm < 2; fm++)
#pragma unroll
    for (int r = 0; r < 8; r++)
      Sf[(wm << 5) + (fm << 4) + r + (hi << 3)][(wn << 4) + l15] = s[fm][r] * tv;
  __syncthreads();

  if (tid < 64) {  // row softmax
    float mx = -1e30f;
    for (int j = 0; j < 64; j++) mx = fmaxf(mx, Sf[tid][j]);
    float sum = 0.f;
    for (int j = 0; j < 64; j++) sum += expf(Sf[tid][j] - mx);
    float inv = 1.0f / sum;
    for (int j = 0; j < 64; j++) Pb[tid][j] = (__bf16)(expf(Sf[tid][j] - mx) * inv);
  }
  __syncthreads();

  const __bf16* vbase = vb + (size_t)b * 1024 * 512 + h * 64;
  __bf16* obase = ob + (size_t)b * 1024 * 512 + h * 64;
  for (int nb = wave; nb < 64; nb += 8) {  // uniform per wave: EXEC all ones
    v8f o[4];
#pragma unroll
    for (int i = 0; i < 4; i++)
#pragma unroll
      for (int r = 0; r < 8; r++) o[i][r] = 0.0f;
#pragma unroll
    for (int kk = 0; kk < 64; kk += 32) {
      Frag bf;
      int n_ = (nb << 4) + l15;
      bf.u[0] = *(const uint4*)(vbase + (size_t)n_ * 512 + kk + (hi << 3));
      bf.u[1] = *(const uint4*)(vbase + (size_t)n_ * 512 + kk + 16 + (hi << 3));
#pragma unroll
      for (int fm = 0; fm < 4; fm++) {
        Frag a;
        int m_ = (fm << 4) + l15;
        a.u[0] = *(const uint4*)&Pb[m_][kk + (hi << 3)];
        a.u[1] = *(const uint4*)&Pb[m_][kk + 16 + (hi << 3)];
        o[fm] = wmma_bf16(a.v, bf.v, o[fm]);
      }
    }
    int n_ = (nb << 4) + l15;
#pragma unroll
    for (int fm = 0; fm < 4; fm++)
#pragma unroll
      for (int r = 0; r < 8; r++)
        obase[(size_t)n_ * 512 + (fm << 4) + r + (hi << 3)] = (__bf16)o[fm][r];
  }
}

// ---------------------------------------------------------------------------
// LPI: depthwise 3x3 conv + GELU  (input (B,C,32,32) fp32)
// ---------------------------------------------------------------------------
__global__ void k_dwconv1_gelu(const float* __restrict__ yln,
                               const float* __restrict__ w,
                               const float* __restrict__ bias,
                               float* __restrict__ t1) {
  int i = blockIdx.x * 256 + threadIdx.x;  // B*C*1024
  int ww = i & 31, hh = (i >> 5) & 31, c = (i >> 10) & 511, b = i >> 19;
  const float* p = yln + ((size_t)(b * 512 + c) << 10);
  const float* wc = w + c * 9;
  float acc = bias[c];
#pragma unroll
  for (int ki = -1; ki <= 1; ki++) {
    int yy = hh + ki;
    if (yy < 0 || yy > 31) continue;
#pragma unroll
    for (int kj = -1; kj <= 1; kj++) {
      int xx = ww + kj;
      if (xx < 0 || xx > 31) continue;
      acc += wc[(ki + 1) * 3 + (kj + 1)] * p[(yy << 5) + xx];
    }
  }
  t1[i] = gelu_f(acc);
}

// BN batch stats per channel -> scale/shift
__global__ __launch_bounds__(256) void k_bn_stats(const float* __restrict__ t1,
                                                  const float* __restrict__ g,
                                                  const float* __restrict__ bb,
                                                  float* __restrict__ bnp) {
  int c = blockIdx.x, tid = threadIdx.x;
  float s = 0.f, s2 = 0.f;
  for (int j = tid; j < 16384; j += 256) {
    int b = j >> 10, hw = j & 1023;
    float v = t1[((size_t)(b * 512 + c) << 10) + hw];
    s += v; s2 += v * v;
  }
  __shared__ float r1[256], r2[256];
  r1[tid] = s; r2[tid] = s2;
  __syncthreads();
  for (int st = 128; st > 0; st >>= 1) {
    if (tid < st) { r1[tid] += r1[tid + st]; r2[tid] += r2[tid + st]; }
    __syncthreads();
  }
  if (tid == 0) {
    float mean = r1[0] * (1.0f / 16384.0f);
    float var = r2[0] * (1.0f / 16384.0f) - mean * mean;
    float sc = g[c] * rsqrtf(var + 1e-5f);
    bnp[c] = sc;
    bnp[512 + c] = bb[c] - mean * sc;
  }
}

// BN(applied inline) -> depthwise conv2 -> + residual x2 ; writes x3 (B,N,C)
__global__ void k_dwconv2_res(const float* __restrict__ t1,
                              const float* __restrict__ bnp,
                              const float* __restrict__ w,
                              const float* __restrict__ bias,
                              const float* __restrict__ x2,
                              float* __restrict__ x3) {
  int i = blockIdx.x * 256 + threadIdx.x;
  int ww = i & 31, hh = (i >> 5) & 31, c = (i >> 10) & 511, b = i >> 19;
  const float* p = t1 + ((size_t)(b * 512 + c) << 10);
  const float* wc = w + c * 9;
  float sc = bnp[c], sh = bnp[512 + c];
  float acc = bias[c];
#pragma unroll
  for (int ki = -1; ki <= 1; ki++) {
    int yy = hh + ki;
    if (yy < 0 || yy > 31) continue;
#pragma unroll
    for (int kj = -1; kj <= 1; kj++) {
      int xx = ww + kj;
      if (xx < 0 || xx > 31) continue;
      acc += wc[(ki + 1) * 3 + (kj + 1)] * (sc * p[(yy << 5) + xx] + sh);
    }
  }
  size_t oi = ((size_t)b * 1024 + (hh << 5) + ww) * 512 + c;
  x3[oi] = x2[oi] + acc;
}

// ---------------------------------------------------------------------------
// Host launcher
// ---------------------------------------------------------------------------
extern "C" void kernel_launch(void* const* d_in, const int* in_sizes, int n_in,
                              void* d_out, int out_size, void* d_ws, size_t ws_size,
                              hipStream_t stream) {
  (void)in_sizes; (void)n_in; (void)out_size; (void)ws_size;
  const float* x       = (const float*)d_in[0];
  const float* norm1_w = (const float*)d_in[3];
  const float* norm1_b = (const float*)d_in[4];
  const float* qkv_w   = (const float*)d_in[5];
  const float* temp    = (const float*)d_in[6];
  const float* proj_w  = (const float*)d_in[7];
  const float* proj_b  = (const float*)d_in[8];
  const float* norm3_w = (const float*)d_in[9];
  const float* norm3_b = (const float*)d_in[10];
  const float* conv1_w = (const float*)d_in[11];
  const float* conv1_b = (const float*)d_in[12];
  const float* bn_g    = (const float*)d_in[13];
  const float* bn_b    = (const float*)d_in[14];
  const float* conv2_w = (const float*)d_in[15];
  const float* conv2_b = (const float*)d_in[16];
  const float* norm2_w = (const float*)d_in[17];
  const float* norm2_b = (const float*)d_in[18];
  const float* fc1_w   = (const float*)d_in[19];
  const float* fc1_b   = (const float*)d_in[20];
  const float* fc2_w   = (const float*)d_in[21];
  const float* fc2_b   = (const float*)d_in[22];

  const int M = 16384;  // B*N

  char* W = (char*)d_ws;
  size_t off = 0;
  auto take = [&](size_t bytes) {
    char* p = W + off;
    off += (bytes + 255) & ~(size_t)255;
    return p;
  };
  __bf16* WQKV = (__bf16*)take((size_t)1536 * 512 * 2);
  __bf16* WPRJ = (__bf16*)take((size_t)512 * 512 * 2);
  __bf16* WFC1 = (__bf16*)take((size_t)2048 * 512 * 2);
  __bf16* WFC2 = (__bf16*)take((size_t)512 * 2048 * 2);
  __bf16* YB   = (__bf16*)take((size_t)M * 512 * 2);      // LN1 out; reused as LN2 out
  char*   QKVR = take((size_t)M * 1536 * 4);              // big region, aliased below
  float*  QKV  = (float*)QKVR;
  float*  YLP  = (float*)QKVR;                            // LN3 transposed (after qkv dead)
  float*  T1   = (float*)(QKVR + (size_t)M * 512 * 4);    // conv1 output
  __bf16* HB   = (__bf16*)QKVR;                           // fc1 activation (after LPI dead)
  __bf16* QH   = (__bf16*)take((size_t)128 * 64 * 1024 * 2);
  __bf16* KH   = (__bf16*)take((size_t)128 * 64 * 1024 * 2);
  __bf16* VB   = (__bf16*)take((size_t)M * 512 * 2);
  __bf16* OB   = (__bf16*)take((size_t)M * 512 * 2);
  float*  X2   = (float*)take((size_t)M * 512 * 4);
  float*  X3   = (float*)take((size_t)M * 512 * 4);
  float*  BNP  = (float*)take((size_t)1024 * 4);

  // weights -> bf16
  k_f2b<<<(1536 * 512) / 256, 256, 0, stream>>>(qkv_w, WQKV, 1536 * 512);
  k_f2b<<<(512 * 512) / 256, 256, 0, stream>>>(proj_w, WPRJ, 512 * 512);
  k_f2b<<<(2048 * 512) / 256, 256, 0, stream>>>(fc1_w, WFC1, 2048 * 512);
  k_f2b<<<(512 * 2048) / 256, 256, 0, stream>>>(fc2_w, WFC2, 512 * 2048);

  // --- attention branch ---
  k_layernorm<0><<<M, 256, 0, stream>>>(x, norm1_w, norm1_b, YB);
  k_gemm_nt<false, false, false, true, false><<<dim3(12, 128), 256, 0, stream>>>(
      YB, WQKV, nullptr, nullptr, QKV, nullptr, M, 1536, 512);
  k_qk_split<<<8192, 128, 0, stream>>>(QKV, QH, KH);
  k_v_split<<<(M * 512) / 256, 256, 0, stream>>>(QKV, VB);
  k_attn<<<128, 256, 0, stream>>>(QH, KH, VB, temp, OB);
  k_gemm_nt<true, false, true, true, false><<<dim3(4, 128), 256, 0, stream>>>(
      OB, WPRJ, proj_b, x, X2, nullptr, M, 512, 512);

  // --- LPI branch ---
  k_layernorm<1><<<M, 256, 0, stream>>>(X2, norm3_w, norm3_b, YLP);
  k_dwconv1_gelu<<<(M * 512) / 256, 256, 0, stream>>>(YLP, conv1_w, conv1_b, T1);
  k_bn_stats<<<512, 256, 0, stream>>>(T1, bn_g, bn_b, BNP);
  k_dwconv2_res<<<(M * 512) / 256, 256, 0, stream>>>(T1, BNP, conv2_w, conv2_b, X2, X3);

  // --- MLP ---
  k_layernorm<0><<<M, 256, 0, stream>>>(X3, norm2_w, norm2_b, YB);
  k_gemm_nt<true, true, false, false, true><<<dim3(16, 128), 256, 0, stream>>>(
      YB, WFC1, fc1_b, nullptr, nullptr, HB, M, 2048, 512);
  k_gemm_nt<true, false, true, true, false><<<dim3(4, 128), 256, 0, stream>>>(
      HB, WFC2, fc2_b, X3, (float*)d_out, nullptr, M, 512, 2048);
}
